// RNN_24240795419042
// MI455X (gfx1250) — matrix operand order repaired and checked by reading, not compiled, not run on hardware
//
#include <hip/hip_runtime.h>
#include <hip/hip_bf16.h>

// ---------------------------------------------------------------------------
// RNN: 128 sequential layers of H = tanh(H@W_hh + X_i@W_hx + b_hh + b_hx),
// then out = H@O_w + O_b.
// Per-layer fused bf16-WMMA GEMM (f32 accumulate), f32 H ping-pong in
// workspace (L2-resident), weights streamed f32 from HBM, converted to bf16
// with packed v_cvt_pk_bf16_f32, double-buffered LDS, 1 barrier per K-step.
// Block: 128 threads = 4 waves (2M x 2N), block tile 64 x 128,
// wave tile 32 x 64 -> 8x v_wmma_f32_16x16x32_bf16 per 32-wide K step.
// ---------------------------------------------------------------------------

#define BATCH   256
#define INPUT   512
#define HIDDEN  1024
#define LAYERS  128
#define OUTDIM  512

typedef __attribute__((ext_vector_type(16))) __bf16 v16bf;
typedef __attribute__((ext_vector_type(8)))  __bf16 v8bf;
typedef __attribute__((ext_vector_type(8)))  float  v8f;

union FragU { uint4 q[2]; v16bf v; };   // 32 bytes = one A/B fragment
union Cvt8  { v8bf v; uint4 q; };       // 8 bf16 = 16 bytes

__global__ void zero_kernel(float* __restrict__ p, int n) {
    int i = blockIdx.x * blockDim.x + threadIdx.x;
    if (i < n) p[i] = 0.0f;
}

// C[256][N] = act( A_cat[256][K] @ B_cat[K][N] + bias1 + bias2 )
//   A rows 0..1023   : A1 = H  (f32, row-major [256][1024])
//   A rows 1024..1535: A2 = X slice (element stride LAYERS in k, INPUT*LAYERS in m)
//   B rows 0..1023   : B1 ([1024][N] row-major f32)
//   B rows 1024..1535: B2 ([512][N] row-major f32)  (null => K = 1024 only)
__global__ __launch_bounds__(128) void rnn_gemm(
    const float* __restrict__ A1,
    const float* __restrict__ A2,
    const float* __restrict__ B1,
    const float* __restrict__ B2,
    const float* __restrict__ bias1,
    const float* __restrict__ bias2,
    float* __restrict__ Cout,
    int N, int applyTanh)
{
    constexpr int LDSROW = 80;          // 32 bf16 (64B) padded to 80B
    constexpr int ASZ = 64 * LDSROW;    // 5120 B
    constexpr int BSZ = 128 * LDSROW;   // 10240 B
    __shared__ __align__(16) unsigned char ldsA[2 * ASZ];
    __shared__ __align__(16) unsigned char ldsB[2 * BSZ];

    const int tid  = threadIdx.x;
    const int wave = tid >> 5;
    const int lane = tid & 31;
    const int lo   = lane & 15;      // fragment: row/col index within 16
    const int hi   = lane >> 4;      // fragment: K-group select
    const int wm   = wave >> 1;      // 0..1 -> M offset wm*32
    const int wn   = wave & 1;       // 0..1 -> N offset wn*64
    const int m0   = blockIdx.y * 64;
    const int n0   = blockIdx.x * 128;

    const int KT = A2 ? (1024 + INPUT) / 32 : 1024 / 32;   // 48 or 32 K-steps

    // cooperative fill roles (128 threads)
    const int ar = tid >> 1;         // A row 0..63
    const int ac = (tid & 1) * 16;   // A col base 0/16
    const int bc = tid;              // B col 0..127 (transposed row in LDS)

    v8f aReg0, aReg1;                // 16 A floats
    v8f bReg[4];                     // 32 B floats (one full column)

    auto loadTiles = [&](int kt) {
        const int kb = kt * 32;
        // ---- A tile: 64 x 32 f32, 16 per thread ----
        {
            const int m = m0 + ar;
            const int k = kb + ac;
            if (k < 1024) {
                const float4* p = (const float4*)(A1 + (size_t)m * 1024 + k);
                float4 x = p[0], y = p[1], z = p[2], w = p[3];
                aReg0[0]=x.x; aReg0[1]=x.y; aReg0[2]=x.z; aReg0[3]=x.w;
                aReg0[4]=y.x; aReg0[5]=y.y; aReg0[6]=y.z; aReg0[7]=y.w;
                aReg1[0]=z.x; aReg1[1]=z.y; aReg1[2]=z.z; aReg1[3]=z.w;
                aReg1[4]=w.x; aReg1[5]=w.y; aReg1[6]=w.z; aReg1[7]=w.w;
            } else {
                const int kk = k - 1024;
                const float* base = A2 + ((size_t)m * INPUT + kk) * LAYERS;
#pragma unroll
                for (int j = 0; j < 8; ++j) aReg0[j] = base[(size_t)j * LAYERS];
#pragma unroll
                for (int j = 0; j < 8; ++j) aReg1[j] = base[(size_t)(8 + j) * LAYERS];
            }
        }
        // ---- B tile: 32 x 128 f32, one column per thread (coalesced) ----
        {
            const float* src;
            int krow;
            if (kb < 1024) { src = B1; krow = kb; }
            else           { src = B2; krow = kb - 1024; }
            const float* col = src + (size_t)krow * N + n0 + bc;
#pragma unroll
            for (int g = 0; g < 4; ++g)
#pragma unroll
                for (int j = 0; j < 8; ++j)
                    bReg[g][j] = col[(size_t)(g * 8 + j) * N];
        }
    };

    auto storeTiles = [&](int buf) {
        unsigned char* aDst = ldsA + buf * ASZ + ar * LDSROW + ac * 2;
        Cvt8 c0, c1;
        c0.v = __builtin_convertvector(aReg0, v8bf);
        c1.v = __builtin_convertvector(aReg1, v8bf);
        ((uint4*)aDst)[0] = c0.q;
        ((uint4*)aDst)[1] = c1.q;

        uint4* bDst = (uint4*)(ldsB + buf * BSZ + bc * LDSROW);  // 16B aligned
#pragma unroll
        for (int g = 0; g < 4; ++g) {
            Cvt8 cb;
            cb.v = __builtin_convertvector(bReg[g], v8bf);
            bDst[g] = cb.q;                       // k = g*8 .. g*8+7
        }
    };

    v8f acc[2][4];
#pragma unroll
    for (int i = 0; i < 2; ++i)
#pragma unroll
        for (int j = 0; j < 4; ++j) {
            v8f z = {0.f, 0.f, 0.f, 0.f, 0.f, 0.f, 0.f, 0.f};
            acc[i][j] = z;
        }

    loadTiles(0);
    storeTiles(0);
    for (int kt = 0; kt < KT; ++kt) {
        __syncthreads();                       // buf[kt&1] ready for reads
        if (kt + 1 < KT) loadTiles(kt + 1);    // global loads overlap WMMAs
        if (kt + 2 < KT) {                     // prefetch weight stream k+2
            const int kpf = (kt + 2) * 32 + (tid >> 2);
            const float* row = (kpf < 1024)
                                   ? (B1 + (size_t)kpf * N)
                                   : (B2 + (size_t)(kpf - 1024) * N);
            __builtin_prefetch(row + n0 + (tid & 3) * 32, 0, 1);
        }

        // fragment loads (2x ds_load_b128 each) + 8 WMMAs
        const int buf = kt & 1;
        FragU afr[2], bfr[4];
#pragma unroll
        for (int tm = 0; tm < 2; ++tm) {
            const unsigned char* base =
                ldsA + buf * ASZ + (wm * 32 + tm * 16 + lo) * LDSROW + hi * 16;
            afr[tm].q[0] = *(const uint4*)(base);        // K = hi*8 + 0..7
            afr[tm].q[1] = *(const uint4*)(base + 32);   // K = 16 + hi*8 + 0..7
        }
#pragma unroll
        for (int tn = 0; tn < 4; ++tn) {
            const unsigned char* base =
                ldsB + buf * BSZ + (wn * 64 + tn * 16 + lo) * LDSROW + hi * 16;
            bfr[tn].q[0] = *(const uint4*)(base);        // K = hi*8 + 0..7
            bfr[tn].q[1] = *(const uint4*)(base + 32);   // K = 16 + hi*8 + 0..7
        }
#pragma unroll
        for (int tm = 0; tm < 2; ++tm)
#pragma unroll
            for (int tn = 0; tn < 4; ++tn)
                acc[tm][tn] = __builtin_amdgcn_wmma_f32_16x16x32_bf16(
                    false, afr[tm].v, false, bfr[tn].v,
                    (short)0, acc[tm][tn], false, false);

        if (kt + 1 < KT) storeTiles((kt + 1) & 1);  // other buffer: no 2nd barrier
    }

    // epilogue: bias + tanh + f32 store
#pragma unroll
    for (int tm = 0; tm < 2; ++tm) {
#pragma unroll
        for (int tn = 0; tn < 4; ++tn) {
            const int col = n0 + wn * 64 + tn * 16 + lo;
            float b = bias1[col];
            if (bias2) b += bias2[col];
#pragma unroll
            for (int v = 0; v < 8; ++v) {
                const int row = m0 + wm * 32 + tm * 16 + hi * 8 + v;
                float x = acc[tm][tn][v] + b;
                if (applyTanh) {
                    float e = __expf(2.0f * x);                // v_exp_f32
                    float r = __builtin_amdgcn_rcpf(e + 1.0f); // v_rcp_f32
                    x = 1.0f - 2.0f * r;                       // saturation-safe
                }
                Cout[(size_t)row * N + col] = x;
            }
        }
    }
}

extern "C" void kernel_launch(void* const* d_in, const int* in_sizes, int n_in,
                              void* d_out, int out_size, void* d_ws, size_t ws_size,
                              hipStream_t stream) {
    (void)in_sizes; (void)n_in; (void)out_size; (void)ws_size;
    const float* X    = (const float*)d_in[0];  // [256][512][128]
    const float* W_hx = (const float*)d_in[1];  // [128][512][1024]
    const float* b_hx = (const float*)d_in[2];  // [128][1024]
    const float* W_hh = (const float*)d_in[3];  // [128][1024][1024]
    const float* b_hh = (const float*)d_in[4];  // [128][1024]
    const float* O_w  = (const float*)d_in[5];  // [1024][512]
    const float* O_b  = (const float*)d_in[6];  // [512]
    float* out = (float*)d_out;                 // [256][512]

    float* H0 = (float*)d_ws;
    float* H1 = H0 + (size_t)BATCH * HIDDEN;

    const int hElems = BATCH * HIDDEN;
    zero_kernel<<<(hElems + 255) / 256, 256, 0, stream>>>(H0, hElems);

    for (int i = 0; i < LAYERS; ++i) {
        const float* Hin = (i & 1) ? H1 : H0;
        float*       Hot = (i & 1) ? H0 : H1;
        rnn_gemm<<<dim3(HIDDEN / 128, BATCH / 64), 128, 0, stream>>>(
            Hin,
            X + i,                                   // X[:, :, i], strided
            W_hh + (size_t)i * HIDDEN * HIDDEN,
            W_hx + (size_t)i * INPUT * HIDDEN,
            b_hh + (size_t)i * HIDDEN,
            b_hx + (size_t)i * HIDDEN,
            Hot, HIDDEN, /*tanh=*/1);
    }
    // layer 127 (odd) wrote H0 -> final hidden state is H0
    rnn_gemm<<<dim3(OUTDIM / 128, BATCH / 64), 128, 0, stream>>>(
        H0, nullptr, O_w, nullptr, O_b, nullptr, out, OUTDIM, /*tanh=*/0);
}
